// FrameTokenCoSelector_4028679323664
// MI455X (gfx1250) — matrix operand order repaired and checked by reading, not compiled, not run on hardware
//
#include <hip/hip_runtime.h>
#include <math.h>

typedef __attribute__((ext_vector_type(16))) _Float16 v16h;
typedef __attribute__((ext_vector_type(8)))  _Float16 v8h;
typedef __attribute__((ext_vector_type(8)))  float    v8f;

#define CB   8      // batch
#define CT   32     // frames
#define CN   1024   // tokens per frame
#define CD   256    // feature dim
#define CHID 1024   // frame-MLP hidden
#define CDH  128    // token-MLP hidden
#define CKF  8
#define CKT  128
#define APAD 264    // 256 + 8 f16 pad (16B multiple, bank-friendly)

__device__ inline float gelu_f(float x) {
    return 0.5f * x * (1.0f + erff(x * 0.70710678118654752f));
}

__device__ inline float wave_sum32(float v) {
    #pragma unroll
    for (int m = 16; m >= 1; m >>= 1) v += __shfl_xor(v, m, 32);
    return v;
}

// ---------------------------------------------------------------- K0: weights -> f16
__global__ __launch_bounds__(256) void prep_weights_kernel(
    const float* __restrict__ tk_w1, const float* __restrict__ fr_w1,
    _Float16* __restrict__ tkh, _Float16* __restrict__ frh)
{
    int i = blockIdx.x * 256 + threadIdx.x;
    if (i < CDH * CD) tkh[i] = (_Float16)tk_w1[i];
    int j = i - CDH * CD;
    if (j >= 0 && j < CHID * CD) frh[j] = (_Float16)fr_w1[j];
}

// ---------------------------------------------------------------- K1: frame_repr (masked mean)
__global__ __launch_bounds__(256) void frame_repr_kernel(
    const float* __restrict__ x, const float* __restrict__ mask,
    float* __restrict__ frepr)
{
    int bt = blockIdx.x, d = threadIdx.x;
    const float* xp = x + (long)bt * CN * CD;
    const float* mp = mask + (long)bt * CN;
    float acc = 0.f, ms = 0.f;
    for (int n = 0; n < CN; ++n) {
        float m = mp[n];
        acc += xp[(long)n * CD + d] * m;
        ms  += m;
    }
    frepr[(long)bt * CD + d] = acc / fmaxf(ms, 1e-6f);
}

// ---------------------------------------------------------------- K2: frame MLP (16 rows / block)
__global__ __launch_bounds__(256) void frame_mlp_kernel(
    const float* __restrict__ frepr,
    const float* __restrict__ ln_g, const float* __restrict__ ln_b,
    const _Float16* __restrict__ w1h, const float* __restrict__ b1,
    const float* __restrict__ w2, const float* __restrict__ b2,
    float* __restrict__ flogit_out)
{
    __shared__ __align__(16) _Float16 Abuf[16][APAD];
    __shared__ float flog[16];
    const int tid = threadIdx.x, lane = tid & 31, wave = tid >> 5;
    const int row0 = blockIdx.x * 16;
    if (tid < 16) flog[tid] = 0.f;

    // LayerNorm: each wave normalizes 2 rows, writes f16 A-tile to LDS
    #pragma unroll
    for (int i = 0; i < 2; ++i) {
        int lr = wave * 2 + i;
        const float* xr = frepr + (long)(row0 + lr) * CD + lane * 8;
        float4 u0 = *(const float4*)xr;
        float4 u1 = *(const float4*)(xr + 4);
        float vs[8] = {u0.x,u0.y,u0.z,u0.w,u1.x,u1.y,u1.z,u1.w};
        float s = 0.f, q = 0.f;
        #pragma unroll
        for (int j = 0; j < 8; ++j) { s += vs[j]; q += vs[j]*vs[j]; }
        s = wave_sum32(s); q = wave_sum32(q);
        float mu = s * (1.0f/CD);
        float rstd = rsqrtf(q * (1.0f/CD) - mu*mu + 1e-5f);
        #pragma unroll
        for (int j = 0; j < 8; ++j) {
            int d = lane * 8 + j;
            Abuf[lr][d] = (_Float16)((vs[j]-mu)*rstd*ln_g[d] + ln_b[d]);
        }
    }
    __syncthreads();

    const int l16 = lane & 15, h = lane >> 4;
    const int ntb = wave * 8;           // 8 N-tiles per wave (64 total)
    v8f acc[8] = {};
    for (int kk = 0; kk < 8; ++kk) {
        v8h alo = *(const v8h*)&Abuf[l16][kk*32 + h*16];
        v8h ahi = *(const v8h*)&Abuf[l16][kk*32 + h*16 + 8];
        v16h a = __builtin_shufflevector(alo, ahi, 0,1,2,3,4,5,6,7,8,9,10,11,12,13,14,15);
        #pragma unroll
        for (int j = 0; j < 8; ++j) {
            int nc = (ntb + j) * 16 + l16;
            const _Float16* bp = w1h + (long)nc * CD + kk*32 + h*16;
            v8h blo = *(const v8h*)bp;
            v8h bhi = *(const v8h*)(bp + 8);
            v16h bv = __builtin_shufflevector(blo, bhi, 0,1,2,3,4,5,6,7,8,9,10,11,12,13,14,15);
            acc[j] = __builtin_amdgcn_wmma_f32_16x16x32_f16(
                false, a, false, bv, (short)0, acc[j], false, false);
        }
    }
    // GeLU + dot with w2, reduce per row
    #pragma unroll
    for (int r = 0; r < 8; ++r) {
        float p = 0.f;
        #pragma unroll
        for (int j = 0; j < 8; ++j) {
            int nc = (ntb + j) * 16 + l16;
            p += gelu_f(acc[j][r] + b1[nc]) * w2[nc];
        }
        #pragma unroll
        for (int m = 8; m >= 1; m >>= 1) p += __shfl_xor(p, m, 32);
        if (l16 == 0) atomicAdd(&flog[h*8 + r], p);
    }
    __syncthreads();
    if (tid < 16) flogit_out[row0 + tid] = flog[tid] + b2[0];
}

// ---------------------------------------------------------------- K3: frame softmax + top-8
__global__ __launch_bounds__(32) void frame_topk_kernel(
    const float* __restrict__ flogit, float* __restrict__ fsoft_ws,
    int* __restrict__ fidx_ws, float* __restrict__ fmask_out,
    float* __restrict__ fidx_out)
{
    int b = blockIdx.x, t = threadIdx.x;       // 32 threads = 1 wave
    float l = flogit[b*CT + t];
    float mx = l;
    #pragma unroll
    for (int m = 16; m >= 1; m >>= 1) mx = fmaxf(mx, __shfl_xor(mx, m, 32));
    float e = __expf(l - mx);
    float s = wave_sum32(e);
    float soft = e / s;
    fsoft_ws[b*CT + t] = soft;
    fmask_out[b*CT + t] = soft;                 // forward value of ST mask == soft
    float v = l;
    for (int it = 0; it < CKF; ++it) {
        float bv = v; int bi = t;
        #pragma unroll
        for (int m = 16; m >= 1; m >>= 1) {
            float ov = __shfl_xor(bv, m, 32); int oi = __shfl_xor(bi, m, 32);
            if (ov > bv || (ov == bv && oi < bi)) { bv = ov; bi = oi; }
        }
        if (t == 0) { fidx_ws[b*CKF + it] = bi; fidx_out[b*CKF + it] = (float)bi; }
        if (t == bi) v = -INFINITY;
    }
}

// ---------------------------------------------------------------- K4: token MLP (64 rows / block)
__global__ __launch_bounds__(256) void token_mlp_kernel(
    const float* __restrict__ x, const float* __restrict__ mask,
    const float* __restrict__ ln_g, const float* __restrict__ ln_b,
    const _Float16* __restrict__ w1h, const float* __restrict__ b1,
    const float* __restrict__ w2, const float* __restrict__ b2,
    float* __restrict__ tlogit)
{
    __shared__ __align__(16) _Float16 Abuf[64][APAD];
    __shared__ float flog[64];
    const int tid = threadIdx.x, lane = tid & 31, wave = tid >> 5;
    const long rowbase = (long)blockIdx.x * 64;
    if (tid < 64) flog[tid] = 0.f;

    // LayerNorm: each wave handles 8 rows
    #pragma unroll
    for (int i = 0; i < 8; ++i) {
        int lr = wave * 8 + i;
        const float* xr = x + (rowbase + lr) * CD + lane * 8;
        float4 u0 = *(const float4*)xr;
        float4 u1 = *(const float4*)(xr + 4);
        float vs[8] = {u0.x,u0.y,u0.z,u0.w,u1.x,u1.y,u1.z,u1.w};
        float s = 0.f, q = 0.f;
        #pragma unroll
        for (int j = 0; j < 8; ++j) { s += vs[j]; q += vs[j]*vs[j]; }
        s = wave_sum32(s); q = wave_sum32(q);
        float mu = s * (1.0f/CD);
        float rstd = rsqrtf(q * (1.0f/CD) - mu*mu + 1e-5f);
        #pragma unroll
        for (int j = 0; j < 8; ++j) {
            int d = lane * 8 + j;
            Abuf[lr][d] = (_Float16)((vs[j]-mu)*rstd*ln_g[d] + ln_b[d]);
        }
    }
    __syncthreads();

    // GEMM [64 x 256] x [256 x 128]: wave owns 1 M-tile x 4 N-tiles
    const int mt = wave & 3;
    const int ntb = (wave >> 2) * 4;
    const int l16 = lane & 15, h = lane >> 4;
    const int arow = mt * 16 + l16;
    v8f acc[4] = {};
    for (int kk = 0; kk < 8; ++kk) {
        v8h alo = *(const v8h*)&Abuf[arow][kk*32 + h*16];
        v8h ahi = *(const v8h*)&Abuf[arow][kk*32 + h*16 + 8];
        v16h a = __builtin_shufflevector(alo, ahi, 0,1,2,3,4,5,6,7,8,9,10,11,12,13,14,15);
        #pragma unroll
        for (int j = 0; j < 4; ++j) {
            int nc = (ntb + j) * 16 + l16;
            const _Float16* bp = w1h + (long)nc * CD + kk*32 + h*16;
            v8h blo = *(const v8h*)bp;
            v8h bhi = *(const v8h*)(bp + 8);
            v16h bv = __builtin_shufflevector(blo, bhi, 0,1,2,3,4,5,6,7,8,9,10,11,12,13,14,15);
            acc[j] = __builtin_amdgcn_wmma_f32_16x16x32_f16(
                false, a, false, bv, (short)0, acc[j], false, false);
        }
    }
    #pragma unroll
    for (int r = 0; r < 8; ++r) {
        float p = 0.f;
        #pragma unroll
        for (int j = 0; j < 4; ++j) {
            int nc = (ntb + j) * 16 + l16;
            p += gelu_f(acc[j][r] + b1[nc]) * w2[nc];
        }
        #pragma unroll
        for (int m = 8; m >= 1; m >>= 1) p += __shfl_xor(p, m, 32);
        if (l16 == 0) atomicAdd(&flog[mt*16 + h*8 + r], p);
    }
    __syncthreads();
    if (tid < 64) {
        long grow = rowbase + tid;
        tlogit[grow] = flog[tid] + b2[0] + logf(fmaxf(mask[grow], 1e-9f));
    }
}

// ---------------------------------------------------------------- K5: token softmax + top-128
__global__ __launch_bounds__(256) void token_topk_kernel(
    const float* __restrict__ tlogit, const float* __restrict__ fsoft,
    float* __restrict__ tmask_out, int* __restrict__ tidx_all)
{
    __shared__ float vals[CN];
    __shared__ float rv[256];
    __shared__ int   ri[256];
    int bt = blockIdx.x, tid = threadIdx.x;
    const float* lp = tlogit + (long)bt * CN;
    float myv[4], lm = -INFINITY;
    #pragma unroll
    for (int i = 0; i < 4; ++i) {
        float v = lp[tid + i*256];
        vals[tid + i*256] = v; myv[i] = v; lm = fmaxf(lm, v);
    }
    rv[tid] = lm; __syncthreads();
    for (int s = 128; s >= 1; s >>= 1) { if (tid < s) rv[tid] = fmaxf(rv[tid], rv[tid+s]); __syncthreads(); }
    float gmax = rv[0]; __syncthreads();
    float ls = 0.f;
    #pragma unroll
    for (int i = 0; i < 4; ++i) ls += __expf(myv[i] - gmax);
    rv[tid] = ls; __syncthreads();
    for (int s = 128; s >= 1; s >>= 1) { if (tid < s) rv[tid] += rv[tid+s]; __syncthreads(); }
    float inv = 1.0f / rv[0];
    float fs = fsoft[bt];
    #pragma unroll
    for (int i = 0; i < 4; ++i)
        tmask_out[(long)bt*CN + tid + i*256] = __expf(myv[i] - gmax) * inv * fs;
    __syncthreads();

    for (int it = 0; it < CKT; ++it) {
        float bv = -INFINITY; int bi = 0x7fffffff;
        #pragma unroll
        for (int i = 0; i < 4; ++i) {
            int idx = tid + i*256;
            float v = vals[idx];
            if (v > bv || (v == bv && idx < bi)) { bv = v; bi = idx; }
        }
        rv[tid] = bv; ri[tid] = bi; __syncthreads();
        for (int s = 128; s >= 1; s >>= 1) {
            if (tid < s) {
                float ov = rv[tid+s]; int oi = ri[tid+s];
                if (ov > rv[tid] || (ov == rv[tid] && oi < ri[tid])) { rv[tid] = ov; ri[tid] = oi; }
            }
            __syncthreads();
        }
        if (tid == 0) { tidx_all[bt*CKT + it] = ri[0]; vals[ri[0]] = -INFINITY; }
        __syncthreads();
    }
}

// ---------------------------------------------------------------- K6: gather z, token_idx out
__global__ __launch_bounds__(64) void gather_kernel(
    const float* __restrict__ x, const int* __restrict__ fidx,
    const int* __restrict__ tidx_all, const float* __restrict__ tmask_out,
    float* __restrict__ z, float* __restrict__ tidx_out)
{
    int blk = blockIdx.x;                 // b*KF*KT + f*KT + k
    int b = blk >> 10;
    int rem = blk & 1023;
    int f = rem >> 7;
    int k = rem & 127;
    int fi = fidx[b*CKF + f];
    int ti = tidx_all[(b*CT + fi)*CKT + k];
    long btn = (long)(b*CT + fi)*CN + ti;
    float tm = tmask_out[btn];
    const float4* src = (const float4*)(x + btn * CD);
    float4* dst = (float4*)(z + (long)blk * CD);
    float4 v = src[threadIdx.x];
    v.x *= tm; v.y *= tm; v.z *= tm; v.w *= tm;
    dst[threadIdx.x] = v;
    if (threadIdx.x == 0) tidx_out[blk] = (float)ti;
}

// ================================================================ host launch
extern "C" void kernel_launch(void* const* d_in, const int* in_sizes, int n_in,
                              void* d_out, int out_size, void* d_ws, size_t ws_size,
                              hipStream_t stream) {
    (void)in_sizes; (void)n_in; (void)out_size; (void)ws_size;
    const float* x       = (const float*)d_in[0];
    const float* mask    = (const float*)d_in[1];
    const float* fr_ln_g = (const float*)d_in[2];
    const float* fr_ln_b = (const float*)d_in[3];
    const float* fr_w1   = (const float*)d_in[4];
    const float* fr_b1   = (const float*)d_in[5];
    const float* fr_w2   = (const float*)d_in[6];
    const float* fr_b2   = (const float*)d_in[7];
    const float* tk_ln_g = (const float*)d_in[8];
    const float* tk_ln_b = (const float*)d_in[9];
    const float* tk_w1   = (const float*)d_in[10];
    const float* tk_b1   = (const float*)d_in[11];
    const float* tk_w2   = (const float*)d_in[12];
    const float* tk_b2   = (const float*)d_in[13];

    // workspace layout (bytes)
    char* ws = (char*)d_ws;
    _Float16* tkh      = (_Float16*)(ws + 0);              // 128*256 f16
    _Float16* frh      = (_Float16*)(ws + 65536);          // 1024*256 f16
    float*    frepr    = (float*)   (ws + 589824);         // 256*256 f32
    float*    flogit   = (float*)   (ws + 851968);         // 256 f32
    float*    fsoft    = (float*)   (ws + 852992);         // 256 f32
    int*      fidx     = (int*)     (ws + 854016);         // 64 i32
    float*    tlogit   = (float*)   (ws + 854528);         // 262144 f32
    int*      tidx_all = (int*)     (ws + 1903104);        // 32768 i32

    // output layout (floats)
    float* out = (float*)d_out;
    float* z_out     = out;                  // [8,8,128,256]
    float* fidx_out  = out + 2097152;        // [8,8]
    float* tidx_out  = out + 2097216;        // [8,8,128]
    float* fmask_out = out + 2105408;        // [8,32]
    float* tmask_out = out + 2105664;        // [8,32,1024]

    prep_weights_kernel<<<1152, 256, 0, stream>>>(tk_w1, fr_w1, tkh, frh);
    frame_repr_kernel<<<CB*CT, 256, 0, stream>>>(x, mask, frepr);
    frame_mlp_kernel<<<(CB*CT)/16, 256, 0, stream>>>(
        frepr, fr_ln_g, fr_ln_b, frh, fr_b1, fr_w2, fr_b2, flogit);
    frame_topk_kernel<<<CB, 32, 0, stream>>>(flogit, fsoft, fidx, fmask_out, fidx_out);
    token_mlp_kernel<<<(CB*CT*CN)/64, 256, 0, stream>>>(
        x, mask, tk_ln_g, tk_ln_b, tkh, tk_b1, tk_w2, tk_b2, tlogit);
    token_topk_kernel<<<CB*CT, 256, 0, stream>>>(tlogit, fsoft, tmask_out, tidx_all);
    gather_kernel<<<CB*CKF*CKT, 64, 0, stream>>>(
        x, fidx, tidx_all, tmask_out, z_out, tidx_out);
}
